// Stacked_LSTM_2843268350651
// MI455X (gfx1250) — matrix operand order repaired and checked by reading, not compiled, not run on hardware
//
#include <hip/hip_runtime.h>
#include <hip/hip_bf16.h>

// ---------------------------------------------------------------------------
// Stacked 2-layer LSTM (B=128, T=512, F=64, H=1024) for MI455X / gfx1250.
//
//  * 512 sequential steps -> one persistent kernel; 2 atomic grid barriers per
//    step separate layer1 -> layer2 -> output head.
//  * bf16 weights/activations, fp32 accumulate: v_wmma_f32_16x16x32_bf16.
//    Weights (~25 MB bf16) stay hot in the 192 MB L2.
//  * 64 blocks x 256 threads (8 wave32).  Block owns 16 hidden units/layer
//    (64 gate rows [i,f,g,o]x16); wave owns a 16-row batch tile.  Cell state
//    c1/c2 lives in registers for the whole sequence.
//  * Weight tiles staged into LDS once per block via
//    GLOBAL_LOAD_ASYNC_TO_LDS_B128 (ASYNCcnt, double-buffered 8-kstep chunks,
//    80B padded rows to dodge bank conflicts); waves read B fragments with
//    ds_load_b128.  Removes the 8x redundant weight fetch per block and
//    overlaps the next chunk's DMA with the current chunk's WMMAs.
//  * WMMA fragment layouts as loaded here (contiguous k, same order for A and B):
//      A 16x32 bf16 : lane<16 -> M=lane,   K=k0..k0+15
//                     lane>=16-> M=lane-16,K=k0+16..k0+31
//      B 32x16 bf16 : lane<16 -> N=lane,   K=k0..k0+15   (contig. W row)
//                     lane>=16-> N=lane-16,K=k0+16..k0+31
//      C/D 16x16 f32: VGPR r, lane l -> M=r+8*(l>=16), N=l%16
// ---------------------------------------------------------------------------

typedef __bf16 bf16_t;
typedef __attribute__((ext_vector_type(16))) __bf16 v16bf;
typedef __attribute__((ext_vector_type(8)))  __bf16 v8bf;
typedef __attribute__((ext_vector_type(8)))  float  v8f;
typedef __attribute__((ext_vector_type(4)))  int    v4i;

// Address-space-qualified 16B vector pointers for the async-to-LDS builtin
// (signature: (v4i addrspace(1)*, v4i addrspace(3)*, imm offset, imm cpol)).
typedef __attribute__((address_space(1))) v4i gas_v4i;
typedef __attribute__((address_space(3))) v4i las_v4i;

#define NB     64      // persistent workgroups
#define TPB    256     // 8 wave32 per workgroup
#define BATCH  128
#define SEQ    512
#define FIN    64
#define H      1024
#define HB     16      // hidden units owned per block (NB*HB == H)

// LDS staging geometry: chunk = 8 ksteps x 4 gates; region = 16 rows x 64B
// data padded to 80B row stride (conflict-free 16-lane bank classes).
#define ROWB    80                       // padded row stride (bytes)
#define REGB    (16 * ROWB)              // 1280 B per (kstep,gate) region
#define CHUNKB  (32 * REGB)              // 40960 B per chunk
#define CHUNKE  (CHUNKB / 2)             // elements per chunk

#if defined(__has_builtin)
#if __has_builtin(__builtin_amdgcn_global_load_async_to_lds_b128)
#define ASYNC_G2L 1
#endif
#endif
#ifndef ASYNC_G2L
#define ASYNC_G2L 0
#endif

__device__ __forceinline__ float sigf(float x) {
  return 1.0f / (1.0f + __expf(-x));
}

// 16B global -> LDS copy (async DMA when available).
__device__ __forceinline__ void cp16_g2s(const bf16_t* __restrict__ g, bf16_t* s) {
#if ASYNC_G2L
  __builtin_amdgcn_global_load_async_to_lds_b128(
      (gas_v4i*)(v4i*)g, (las_v4i*)(v4i*)s, 0, 0);
#else
  *(v4i*)s = *(const v4i*)g;   // global_load_b128 + ds_store_b128 fallback
#endif
}

template <int N>
__device__ __forceinline__ void wait_async() {
#if ASYNC_G2L
#if __has_builtin(__builtin_amdgcn_s_wait_asynccnt)
  __builtin_amdgcn_s_wait_asynccnt(N);
#else
  asm volatile("s_wait_asynccnt %0" ::"n"(N) : "memory");
#endif
#endif
}

// A-fragment loader: rowp points at element k0 of this lane's activation row.
__device__ __forceinline__ v16bf load_a(const bf16_t* __restrict__ rowp, int kshA) {
  v8bf lo = *(const v8bf*)(rowp + kshA);
  v8bf hi = *(const v8bf*)(rowp + kshA + 8);
  return __builtin_shufflevector(lo, hi, 0,1,2,3,4,5,6,7,8,9,10,11,12,13,14,15);
}

// B-fragment loader from LDS region (two 16B ds loads, 16B-aligned).
__device__ __forceinline__ v16bf load_b_lds(const bf16_t* __restrict__ bp) {
  v8bf lo = *(const v8bf*)bp;
  v8bf hi = *(const v8bf*)(bp + 8);
  return __builtin_shufflevector(lo, hi, 0,1,2,3,4,5,6,7,8,9,10,11,12,13,14,15);
}

// Grid-wide barrier: monotonic counter, each block arrives once per phase.
__device__ __forceinline__ void grid_barrier(unsigned* __restrict__ cnt) {
  __syncthreads();
  if (threadIdx.x == 0) {
    __threadfence();
    unsigned prev = __hip_atomic_fetch_add(cnt, 1u, __ATOMIC_ACQ_REL,
                                           __HIP_MEMORY_SCOPE_AGENT);
    unsigned target = (prev / NB + 1u) * (unsigned)NB;
    while (__hip_atomic_load(cnt, __ATOMIC_ACQUIRE,
                             __HIP_MEMORY_SCOPE_AGENT) < target) {
      __builtin_amdgcn_s_sleep(2);
    }
  }
  __syncthreads();
}

// Stage one 8-kstep weight chunk (4 gates x 16 rows x 64B) into LDS.
// 2048 16B data units; each of the 256 threads copies 8.
__device__ __forceinline__ void stage_chunk(const bf16_t* __restrict__ W, int j0,
                                            int kbase, bf16_t* __restrict__ dst) {
  const int tid = threadIdx.x;
#pragma unroll
  for (int u = 0; u < 8; ++u) {
    const int d   = tid * 8 + u;       // data-unit index
    const int seg = d & 3;             // 16B segment within 64B row piece
    const int n   = (d >> 2) & 15;     // B column (hidden unit within block)
    const int g   = (d >> 6) & 3;      // gate
    const int ksl = d >> 8;            // kstep within chunk
    const bf16_t* src = W + (size_t)(g * H + j0 + n) * H + kbase + ksl * 32 + seg * 8;
    bf16_t* dstp = (bf16_t*)((char*)dst + (ksl * 4 + g) * REGB + n * ROWB + seg * 16);
    cp16_g2s(src, dstp);
  }
}

// Gate GEMM with LDS-staged, double-buffered weights.  krow == H.
// NCH chunks of 8 ksteps (NCH*8*32 == K).  All 256 threads participate.
template <int NCH>
__device__ __forceinline__ void gemm_gates_lds(v8f acc[4],
                                               const bf16_t* __restrict__ arow,
                                               const bf16_t* __restrict__ W,
                                               int j0, int kshA, int l16, int lhalf,
                                               bf16_t* __restrict__ smem) {
  stage_chunk(W, j0, 0, smem);                    // chunk 0 -> buf 0
  for (int c = 0; c < NCH; ++c) {
    if (c + 1 < NCH)
      stage_chunk(W, j0, (c + 1) * 256, smem + ((c + 1) & 1) * CHUNKE);
    if (c + 1 < NCH) wait_async<8>(); else wait_async<0>();
    __syncthreads();                              // chunk c resident
    const bf16_t* buf = smem + (c & 1) * CHUNKE;
#pragma unroll
    for (int ksl = 0; ksl < 8; ++ksl) {
      v16bf a = load_a(arow + (c * 8 + ksl) * 32, kshA);
#pragma unroll
      for (int g = 0; g < 4; ++g) {
        const bf16_t* bp = (const bf16_t*)((const char*)buf +
            (ksl * 4 + g) * REGB + l16 * ROWB + lhalf * 32);
        v16bf b = load_b_lds(bp);
        acc[g] = __builtin_amdgcn_wmma_f32_16x16x32_bf16(
            false, a, false, b, (short)0, acc[g], false, false);
      }
    }
    __syncthreads();                              // done reading before reuse
  }
}

// Direct-global gate GEMM (used for the small K=64 x-projection).
__device__ __forceinline__ void gemm_gates(v8f acc[4],
                                           const bf16_t* __restrict__ arow,
                                           const bf16_t* __restrict__ wbase,
                                           int krow, int nk, int kshA) {
#pragma unroll 4
  for (int ks = 0; ks < nk; ++ks) {
    v16bf a = load_a(arow + ks * 32, kshA);
#pragma unroll
    for (int g = 0; g < 4; ++g) {
      v16bf b = *(const v16bf*)(wbase + (size_t)g * H * krow + (size_t)ks * 32);
      acc[g] = __builtin_amdgcn_wmma_f32_16x16x32_bf16(
          false, a, false, b, (short)0, acc[g], false, false);
    }
  }
}

// LSTM pointwise: gates [i,f,g,o] -> update c, store bf16 h.
__device__ __forceinline__ void lstm_pointwise_store(const v8f acc[4], float c[8],
                                                     bf16_t* __restrict__ hnew,
                                                     int mD, int jcol) {
#pragma unroll
  for (int r = 0; r < 8; ++r) {
    float ig = sigf(acc[0][r]);
    float fg = sigf(acc[1][r]);
    float gg = tanhf(acc[2][r]);
    float og = sigf(acc[3][r]);
    c[r] = fg * c[r] + ig * gg;
    hnew[(size_t)(mD + r) * H + jcol] = (bf16_t)(og * tanhf(c[r]));
  }
}

__global__ __launch_bounds__(TPB, 1)
void lstm_stack_kernel(const bf16_t* __restrict__ xbf,
                       const bf16_t* __restrict__ Wih1,
                       const bf16_t* __restrict__ Whh1,
                       const bf16_t* __restrict__ Wih2,
                       const bf16_t* __restrict__ Whh2,
                       const bf16_t* __restrict__ Wlin,
                       const float*  __restrict__ b1,
                       const float*  __restrict__ b2,
                       const float*  __restrict__ blin,
                       bf16_t* __restrict__ h1buf,
                       bf16_t* __restrict__ h2buf,
                       unsigned* __restrict__ bar,
                       float* __restrict__ out) {
  __shared__ __align__(16) bf16_t smem[2 * CHUNKE];   // 80 KB double buffer

  const int blk   = blockIdx.x;
  const int wv    = threadIdx.x >> 5;
  const int lane  = threadIdx.x & 31;
  const int l16   = lane & 15;
  const int lhalf = lane >> 4;
  const int kshA  = lhalf * 16;
  const int kshB  = lhalf * 16;
  const int j0    = blk * HB;
  const int jcol  = j0 + l16;
  const int mrow  = 16 * wv + l16;
  const int mD    = 16 * wv + 8 * lhalf;

  const bf16_t* wih1 = Wih1 + (size_t)jcol * FIN + kshB;

  float bias1[4], bias2[4];
#pragma unroll
  for (int g = 0; g < 4; ++g) {
    bias1[g] = b1[g * H + jcol];
    bias2[g] = b2[g * H + jcol];
  }

  // Output head: blocks 0..31, wave 0, one 16x16 tile each (8 M x 4 N tiles).
  const int omt = blk >> 2, ont = blk & 3;
  const bf16_t* wlin = Wlin + (size_t)(ont * 16 + l16) * H + kshB;
  const float blv = blin[ont * 16 + l16];

  float c1[8], c2[8];
#pragma unroll
  for (int r = 0; r < 8; ++r) { c1[r] = 0.f; c2[r] = 0.f; }

  for (int t = 0; t < SEQ; ++t) {
    const int ri = t & 1, wi = ri ^ 1;
    const bf16_t* h1_old = h1buf + (size_t)ri * (BATCH * H);
    bf16_t*       h1_new = h1buf + (size_t)wi * (BATCH * H);
    const bf16_t* h2_old = h2buf + (size_t)ri * (BATCH * H);
    bf16_t*       h2_new = h2buf + (size_t)wi * (BATCH * H);

    // ---- layer 1: gates = x_t @ Wih1^T + h1 @ Whh1^T + (b_ih1+b_hh1) ----
    v8f acc[4];
#pragma unroll
    for (int g = 0; g < 4; ++g)
#pragma unroll
      for (int r = 0; r < 8; ++r) acc[g][r] = bias1[g];

    gemm_gates(acc, xbf + ((size_t)mrow * SEQ + t) * FIN, wih1, FIN, 2, kshA);
    gemm_gates_lds<4>(acc, h1_old + (size_t)mrow * H, Whh1, j0, kshA, l16, lhalf, smem);
    lstm_pointwise_store(acc, c1, h1_new, mD, jcol);

    grid_barrier(bar);   // layer2 needs the full h1_new vector

    // ---- layer 2 ----
#pragma unroll
    for (int g = 0; g < 4; ++g)
#pragma unroll
      for (int r = 0; r < 8; ++r) acc[g][r] = bias2[g];

    gemm_gates_lds<4>(acc, h1_new + (size_t)mrow * H, Wih2, j0, kshA, l16, lhalf, smem);
    gemm_gates_lds<4>(acc, h2_old + (size_t)mrow * H, Whh2, j0, kshA, l16, lhalf, smem);
    lstm_pointwise_store(acc, c2, h2_new, mD, jcol);

    grid_barrier(bar);   // output head + next step need the full h2_new

    // ---- output head: out_t = h2_new @ Wlin^T + b_lin ----
    if (blk < 32 && wv == 0) {
      v8f oacc;
#pragma unroll
      for (int r = 0; r < 8; ++r) oacc[r] = blv;
      const bf16_t* hrow = h2_new + (size_t)(16 * omt + l16) * H;
#pragma unroll 4
      for (int ks = 0; ks < 32; ++ks) {
        v16bf a = load_a(hrow + ks * 32, kshA);
        v16bf b = *(const v16bf*)(wlin + ks * 32);
        oacc = __builtin_amdgcn_wmma_f32_16x16x32_bf16(
            false, a, false, b, (short)0, oacc, false, false);
      }
#pragma unroll
      for (int r = 0; r < 8; ++r)
        out[(size_t)(16 * omt + 8 * lhalf + r) * (SEQ * FIN) +
            (size_t)t * FIN + ont * 16 + l16] = oacc[r];
    }
  }
}

// ------------------------- preprocessing kernels ---------------------------

__global__ void cvt_f32_to_bf16(const float* __restrict__ s,
                                bf16_t* __restrict__ d, int n) {
  int i = blockIdx.x * blockDim.x + threadIdx.x;
  if (i < n) d[i] = (bf16_t)s[i];
}

__global__ void add_bias_f32(const float* __restrict__ a,
                             const float* __restrict__ b,
                             float* __restrict__ d, int n) {
  int i = blockIdx.x * blockDim.x + threadIdx.x;
  if (i < n) d[i] = a[i] + b[i];
}

__global__ void init_state(bf16_t* __restrict__ h1, bf16_t* __restrict__ h2,
                           unsigned* __restrict__ bar, int n) {
  int i = blockIdx.x * blockDim.x + threadIdx.x;
  if (i == 0) *bar = 0u;
  if (i < n) { h1[i] = (bf16_t)0.f; h2[i] = (bf16_t)0.f; }
}

// ------------------------------- launcher ----------------------------------

extern "C" void kernel_launch(void* const* d_in, const int* in_sizes, int n_in,
                              void* d_out, int out_size, void* d_ws, size_t ws_size,
                              hipStream_t stream) {
  (void)in_sizes; (void)n_in; (void)out_size; (void)ws_size;

  const float* input_seq = (const float*)d_in[0];
  const float* W_ih1 = (const float*)d_in[1];
  const float* W_hh1 = (const float*)d_in[2];
  const float* b_ih1 = (const float*)d_in[3];
  const float* b_hh1 = (const float*)d_in[4];
  const float* W_ih2 = (const float*)d_in[5];
  const float* W_hh2 = (const float*)d_in[6];
  const float* b_ih2 = (const float*)d_in[7];
  const float* b_hh2 = (const float*)d_in[8];
  const float* W_lin = (const float*)d_in[9];
  const float* b_lin = (const float*)d_in[10];

  // Workspace carve-up (~35 MB total), 256-byte aligned slices.
  char* ws = (char*)d_ws;
  size_t off = 0;
  auto alloc = [&](size_t bytes) -> char* {
    char* p = ws + off;
    off += (bytes + 255) & ~(size_t)255;
    return p;
  };
  unsigned* bar = (unsigned*)alloc(256);
  bf16_t* h1buf = (bf16_t*)alloc(2ull * BATCH * H * sizeof(bf16_t));
  bf16_t* h2buf = (bf16_t*)alloc(2ull * BATCH * H * sizeof(bf16_t));
  bf16_t* xbf   = (bf16_t*)alloc((size_t)BATCH * SEQ * FIN * sizeof(bf16_t));
  bf16_t* wih1  = (bf16_t*)alloc((size_t)4 * H * FIN * sizeof(bf16_t));
  bf16_t* whh1  = (bf16_t*)alloc((size_t)4 * H * H * sizeof(bf16_t));
  bf16_t* wih2  = (bf16_t*)alloc((size_t)4 * H * H * sizeof(bf16_t));
  bf16_t* whh2  = (bf16_t*)alloc((size_t)4 * H * H * sizeof(bf16_t));
  bf16_t* wlin  = (bf16_t*)alloc((size_t)FIN * H * sizeof(bf16_t));
  float*  b1    = (float*)alloc((size_t)4 * H * sizeof(float));
  float*  b2    = (float*)alloc((size_t)4 * H * sizeof(float));

  auto cvt = [&](const float* s, bf16_t* d, int n) {
    cvt_f32_to_bf16<<<(n + 255) / 256, 256, 0, stream>>>(s, d, n);
  };
  cvt(input_seq, xbf, BATCH * SEQ * FIN);
  cvt(W_ih1, wih1, 4 * H * FIN);
  cvt(W_hh1, whh1, 4 * H * H);
  cvt(W_ih2, wih2, 4 * H * H);
  cvt(W_hh2, whh2, 4 * H * H);
  cvt(W_lin, wlin, FIN * H);
  add_bias_f32<<<(4 * H + 255) / 256, 256, 0, stream>>>(b_ih1, b_hh1, b1, 4 * H);
  add_bias_f32<<<(4 * H + 255) / 256, 256, 0, stream>>>(b_ih2, b_hh2, b2, 4 * H);
  init_state<<<(2 * BATCH * H + 255) / 256, 256, 0, stream>>>(h1buf, h2buf, bar,
                                                              2 * BATCH * H);

  lstm_stack_kernel<<<NB, TPB, 0, stream>>>(xbf, wih1, whh1, wih2, whh2, wlin,
                                            b1, b2, b_lin, h1buf, h2buf, bar,
                                            (float*)d_out);
}
